// ExactVecKM_58033598103722
// MI455X (gfx1250) — matrix operand order repaired and checked by reading, not compiled, not run on hardware
//
#include <hip/hip_runtime.h>
#include <hip/hip_bf16.h>

typedef __attribute__((ext_vector_type(16))) _Float16 v16h;
typedef __attribute__((ext_vector_type(8)))  float    v8f;

#define NPTS     8192
#define ENCD     384
#define RADIUS_F 0.12f
#define R2_F     (0.12f * 0.12f)
#define SQRT_D_F 19.5959179423f   // sqrt(384)
#define ETILES   6                // 6 x 16 = 96 enc columns per wave

// ---------------------------------------------------------------------------
// Kernel 1: pA = (pts/r) @ A ; store cos/sin as f16, TRANSPOSED (ENC x N)
// so that the GEMM's B-operand (K-major per lane) is a contiguous 32B load.
// ---------------------------------------------------------------------------
__global__ __launch_bounds__(256)
void vkm_encode_kernel(const float* __restrict__ pts, const float* __restrict__ A,
                       _Float16* __restrict__ cT, _Float16* __restrict__ sT)
{
    const int e = blockIdx.x;
    const int i = blockIdx.y * 256 + threadIdx.x;
    const float a0 = A[e];
    const float a1 = A[ENCD + e];
    const float a2 = A[2 * ENCD + e];
    const float inv_r = 1.0f / RADIUS_F;
    const float p = (pts[i * 3 + 0] * a0 + pts[i * 3 + 1] * a1 + pts[i * 3 + 2] * a2) * inv_r;
    float s, c;
    sincosf(p, &s, &c);
    cT[(size_t)e * NPTS + i] = (_Float16)c;
    sT[(size_t)e * NPTS + i] = (_Float16)s;
}

// ---------------------------------------------------------------------------
// Kernel 2: fused masked GEMM.  Gr = J @ cos, Gi = J @ sin via
// v_wmma_f32_16x16x32_f16.  Staging round: 128 points -> LDS (one float4 per
// thread, no divergence), then 4 unrolled sub-chunks of K=32, each doing
// 24 clause-batched b128 B loads, a J-tile build with 4-wide batched LDS
// reads, and 12 back-to-back WMMAs.  48 WMMAs per barrier pair.
// ---------------------------------------------------------------------------
__global__ __launch_bounds__(128, 1)
void vkm_gemm_kernel(const float* __restrict__ pts,
                     const _Float16* __restrict__ cT,
                     const _Float16* __restrict__ sT,
                     float* __restrict__ out)
{
    __shared__ float4 lds_pts[128];    // {x, y, z, |p|^2} per staged point

    const int tid  = threadIdx.x;
    const int wave = tid >> 5;
    const int lane = tid & 31;
    const int half = lane >> 4;        // 0: lanes 0-15, 1: lanes 16-31
    const int mn   = lane & 15;        // row (for A) / column (for B,D)

    const int i0 = blockIdx.x * 64 + wave * 16;
    const int e0 = blockIdx.y * (16 * ETILES);

    // This lane's output-row point (A-matrix row m = lane & 15).
    const float px = pts[(i0 + mn) * 3 + 0];
    const float py = pts[(i0 + mn) * 3 + 1];
    const float pz = pts[(i0 + mn) * 3 + 2];
    // d2 = ni + nj - 2*dot < r^2  <=>  nj - 2*dot < r^2 - ni
    const float thr = R2_F - (px * px + py * py + pz * pz);

    v8f accR[ETILES] = {};   // Gr accumulators
    v8f accI[ETILES] = {};   // Gi accumulators

    // B-operand base pointers: lane (n=mn, half) reads K = half*16 .. +15
    // contiguously from the transposed f16 arrays.
    const _Float16* cb[ETILES];
    const _Float16* sb[ETILES];
#pragma unroll
    for (int t = 0; t < ETILES; ++t) {
        const size_t col = (size_t)(e0 + 16 * t + mn);
        cb[t] = cT + col * NPTS + half * 16;
        sb[t] = sT + col * NPTS + half * 16;
    }

    for (int j0 = 0; j0 < NPTS; j0 += 128) {
        // ---- stage 128 points (+ squared norms); all threads, no branch ----
        const float qx = pts[(j0 + tid) * 3 + 0];
        const float qy = pts[(j0 + tid) * 3 + 1];
        const float qz = pts[(j0 + tid) * 3 + 2];
        float4 qs;
        qs.x = qx; qs.y = qy; qs.z = qz;
        qs.w = qx * qx + qy * qy + qz * qz;
        __syncthreads();                 // previous round's LDS reads done
        lds_pts[tid] = qs;
        __syncthreads();

#pragma unroll
        for (int jj = 0; jj < 4; ++jj) {
            const int base = jj * 32;

            // ---- issue ALL B loads for this sub-chunk (24 x b128) ----
            v16h bc[ETILES], bs[ETILES];
#pragma unroll
            for (int t = 0; t < ETILES; ++t) {
                bc[t] = *(const v16h*)(cb[t] + j0 + base);   // 32B aligned
                bs[t] = *(const v16h*)(sb[t] + j0 + base);
            }

            // ---- build J tile; LDS reads batched 4-wide ----
            // 16-bit A layout (matches B): elem e -> K = 16*half + e
            v16h aJ;
#pragma unroll
            for (int g = 0; g < 4; ++g) {
                float4 q[4];
#pragma unroll
                for (int u = 0; u < 4; ++u) {
                    const int el = 4 * g + u;
                    const int k  = 16 * half + el;
                    q[u] = lds_pts[base + k];
                }
#pragma unroll
                for (int u = 0; u < 4; ++u) {
                    const int el = 4 * g + u;
                    const float dot = fmaf(px, q[u].x, fmaf(py, q[u].y, pz * q[u].z));
                    const float t2  = fmaf(-2.0f, dot, q[u].w);   // nj - 2*dot
                    aJ[el] = (t2 < thr) ? (_Float16)1.0f : (_Float16)0.0f;
                }
            }

            // ---- 12 WMMAs back-to-back ----
#pragma unroll
            for (int t = 0; t < ETILES; ++t) {
                accR[t] = __builtin_amdgcn_wmma_f32_16x16x32_f16(
                    false, aJ, false, bc[t], (short)0, accR[t], false, false);
                accI[t] = __builtin_amdgcn_wmma_f32_16x16x32_f16(
                    false, aJ, false, bs[t], (short)0, accI[t], false, false);
            }
        }
    }

    // Store interleaved (Gr, Gi). C/D layout: VGPR v, lane -> M = v + 8*half,
    // N = lane & 15.
#pragma unroll
    for (int t = 0; t < ETILES; ++t) {
        const int col = e0 + 16 * t + mn;
#pragma unroll
        for (int v = 0; v < 8; ++v) {
            const int row = i0 + v + 8 * half;
            float2 g;
            g.x = accR[t][v];
            g.y = accI[t][v];
            *(float2*)(out + (size_t)row * (2 * ENCD) + 2 * col) = g;
        }
    }
}

// ---------------------------------------------------------------------------
// Kernel 3: per-row phase division + normalization, in place on d_out.
// ---------------------------------------------------------------------------
__global__ __launch_bounds__(192)
void vkm_finalize_kernel(const float* __restrict__ pts, const float* __restrict__ A,
                         float* __restrict__ out)
{
    __shared__ float red[192];
    const int i   = blockIdx.x;
    const int tid = threadIdx.x;

    const float px = pts[i * 3 + 0];
    const float py = pts[i * 3 + 1];
    const float pz = pts[i * 3 + 2];
    const float inv_r = 1.0f / RADIUS_F;

    float gr[2], gi[2];
    float norm2 = 0.0f;
#pragma unroll
    for (int q = 0; q < 2; ++q) {
        const int e = tid + q * 192;
        const float a0 = A[e];
        const float a1 = A[ENCD + e];
        const float a2 = A[2 * ENCD + e];
        const float p  = (px * a0 + py * a1 + pz * a2) * inv_r;
        float s, c;
        sincosf(p, &s, &c);
        const float2 g = *(const float2*)(out + (size_t)i * (2 * ENCD) + 2 * e);
        const float inv = 1.0f / (c * c + s * s);   // exact complex division
        gr[q] = (g.x * c + g.y * s) * inv;
        gi[q] = (g.y * c - g.x * s) * inv;
        norm2 += gr[q] * gr[q] + gi[q] * gi[q];
    }

    red[tid] = norm2;
    __syncthreads();
    if (tid < 64) red[tid] += red[tid + 64] + red[tid + 128];
    __syncthreads();
    for (int off = 32; off > 0; off >>= 1) {
        if (tid < off) red[tid] += red[tid + off];
        __syncthreads();
    }
    const float scale = SQRT_D_F * rsqrtf(red[0]);

#pragma unroll
    for (int q = 0; q < 2; ++q) {
        const int e = tid + q * 192;
        float2 g;
        g.x = gr[q] * scale;
        g.y = gi[q] * scale;
        *(float2*)(out + (size_t)i * (2 * ENCD) + 2 * e) = g;
    }
}

// ---------------------------------------------------------------------------
extern "C" void kernel_launch(void* const* d_in, const int* in_sizes, int n_in,
                              void* d_out, int out_size, void* d_ws, size_t ws_size,
                              hipStream_t stream)
{
    (void)in_sizes; (void)n_in; (void)out_size; (void)ws_size;

    const float* pts = (const float*)d_in[0];   // (8192, 3)
    const float* A   = (const float*)d_in[1];   // (3, 384)
    float*       out = (float*)d_out;           // (8192, 384) complex -> interleaved f32

    _Float16* cT = (_Float16*)d_ws;             // (384, 8192) f16 transposed cos
    _Float16* sT = cT + (size_t)NPTS * ENCD;    // (384, 8192) f16 transposed sin

    vkm_encode_kernel<<<dim3(ENCD, NPTS / 256), 256, 0, stream>>>(pts, A, cT, sT);
    vkm_gemm_kernel<<<dim3(NPTS / 64, ENCD / (16 * ETILES)), 128, 0, stream>>>(pts, cT, sT, out);
    vkm_finalize_kernel<<<NPTS, 192, 0, stream>>>(pts, A, out);
}
